// CrossAttentionBlock_57930518889013
// MI455X (gfx1250) — compile-verified, matrix operand order checked by reading, not run on hardware
//
#include <hip/hip_runtime.h>

// ---------------- problem constants (match reference) ----------------
#define BB 4
#define SS 2048
#define TT 256
#define HH 2048
#define NHH 16
#define HD 128
#define LN_EPS 1e-5f
#define SCALE_INV 0.08838834764831845f   // 1/sqrt(128)

typedef __attribute__((ext_vector_type(16))) __bf16 v16bf;
typedef __attribute__((ext_vector_type(8)))  float  v8f;

union Frag {
    unsigned int u[8];
    v16bf v;
};

// ---- optional CDNA5 async global->LDS copy path (compile-safe probe) ----
#if defined(__has_builtin)
#if __has_builtin(__builtin_amdgcn_global_load_async_to_lds_b128)
#define HAVE_ASYNC_LDS 1
#endif
#endif
#ifndef HAVE_ASYNC_LDS
#define HAVE_ASYNC_LDS 0
#endif

#if HAVE_ASYNC_LDS
typedef int gv4i __attribute__((vector_size(16)));   // matches builtin param type
static __device__ __forceinline__ void async_cp16(const void* g, void* l) {
    __builtin_amdgcn_global_load_async_to_lds_b128(
        (__attribute__((address_space(1))) gv4i*)g,
        (__attribute__((address_space(3))) gv4i*)l, 0, 0);
}
static __device__ __forceinline__ void wait_async0() {
#if __has_builtin(__builtin_amdgcn_s_wait_asynccnt)
    __builtin_amdgcn_s_wait_asynccnt(0);
#else
    asm volatile("s_wait_asynccnt 0" ::: "memory");
#endif
}
#endif

static __device__ __forceinline__ float n2n(float x) {
    if (isnan(x)) return 0.0f;
    if (isinf(x)) return x > 0.0f ? 1e4f : -1e4f;
    return x;
}

static __device__ __forceinline__ unsigned short f2bf(float f) {
    union { float f; unsigned int u; } c;
    c.f = f;
    unsigned int r = c.u + 0x7fffu + ((c.u >> 16) & 1u);  // RNE
    return (unsigned short)(r >> 16);
}

// ---------------- kernel 1: f32 -> bf16 convert (+nan_to_num) ----------------
__global__ __launch_bounds__(256) void convert_bf16_kernel(const float* __restrict__ in,
                                                           unsigned short* __restrict__ out,
                                                           long n) {
    long i = (long)blockIdx.x * blockDim.x + threadIdx.x;
    long stride = (long)gridDim.x * blockDim.x;
    for (; i < n; i += stride) out[i] = f2bf(n2n(in[i]));
}

// ---------------- kernel 2: tiled bf16 WMMA GEMM  C = A @ W^T (+bias) ------
// Block tile 128x128, K-stage 32, double-buffered LDS, software pipelined.
// 8 waves; each wave computes a 32x64 sub-tile = 2x4 accumulators.
// A: [M,K] bf16 row-major, W: [N,K] bf16 row-major.
// EPI 0: split-heads bf16 -> out[((b*NH+h)*L + s)*HD + d],  b=m/L, s=m%L
// EPI 2: transposed       -> out[((b*NH+h)*HD + d)*L + t],  b=m/L, t=m%L
// EPI 3: f32 out = acc + bias + n2n(resid[m,n]), row-major [M,N]
template <int EPI>
__global__ __launch_bounds__(256) void gemm_bf16_kernel(
    const unsigned short* __restrict__ A,
    const unsigned short* __restrict__ W,
    const float* __restrict__ bias,
    void* __restrict__ out,
    const float* __restrict__ resid,
    int M, int N, int K, int L) {

    __shared__ unsigned short As[2][128 * 32];
    __shared__ unsigned short Bs[2][128 * 32];

    const int tid  = threadIdx.x;
    const int lane = tid & 31;
    const int wid  = tid >> 5;           // 0..7
    const int wave_m = wid >> 1;         // 0..3 -> 32 rows each
    const int wave_n = wid & 1;          // 0..1 -> 64 cols each

    const int m0 = blockIdx.y * 128;
    const int n0 = blockIdx.x * 128;

    // stage loader: 512 uint4 per matrix; thread handles rows lrow0 and lrow0+64
    const int lrow0 = tid >> 2;          // 0..63
    const int lch   = (tid & 3) * 8;     // bf16 offset 0,8,16,24

    const int phi = (lane & 16) ? 4 : 0;
    const int r15 = lane & 15;

    v8f acc[2][4];
#pragma unroll
    for (int i = 0; i < 2; ++i)
#pragma unroll
        for (int j = 0; j < 4; ++j) acc[i][j] = (v8f){};

    const int NK = K >> 5;

    const size_t aRow0 = (size_t)(m0 + lrow0) * K + lch;
    const size_t aRow1 = (size_t)(m0 + lrow0 + 64) * K + lch;
    const size_t bRow0 = (size_t)(n0 + lrow0) * K + lch;
    const size_t bRow1 = (size_t)(n0 + lrow0 + 64) * K + lch;
    const int ldsOff0 = lrow0 * 32 + lch;
    const int ldsOff1 = (lrow0 + 64) * 32 + lch;

#if HAVE_ASYNC_LDS
    // ---- async-to-LDS pipelined path ----
    {
        // prologue: stage 0 into buf 0
        async_cp16(A + aRow0, &As[0][ldsOff0]);
        async_cp16(A + aRow1, &As[0][ldsOff1]);
        async_cp16(W + bRow0, &Bs[0][ldsOff0]);
        async_cp16(W + bRow1, &Bs[0][ldsOff1]);
        wait_async0();
    }
    __syncthreads();
#else
    uint4 ra0, ra1, rb0, rb1;
    {
        ra0 = *(const uint4*)(A + aRow0);
        ra1 = *(const uint4*)(A + aRow1);
        rb0 = *(const uint4*)(W + bRow0);
        rb1 = *(const uint4*)(W + bRow1);
        *(uint4*)(&As[0][ldsOff0]) = ra0;
        *(uint4*)(&As[0][ldsOff1]) = ra1;
        *(uint4*)(&Bs[0][ldsOff0]) = rb0;
        *(uint4*)(&Bs[0][ldsOff1]) = rb1;
    }
    __syncthreads();
#endif

    int buf = 0;
    for (int i = 0; i < NK; ++i) {
        const int nbuf = buf ^ 1;
        const bool more = (i + 1 < NK);
        const int kn = (i + 1) << 5;

        // ---- issue next stage ----
#if !HAVE_ASYNC_LDS
        uint4 ra0, ra1, rb0, rb1;
#endif
        if (more) {
#if HAVE_ASYNC_LDS
            async_cp16(A + aRow0 + kn, &As[nbuf][ldsOff0]);
            async_cp16(A + aRow1 + kn, &As[nbuf][ldsOff1]);
            async_cp16(W + bRow0 + kn, &Bs[nbuf][ldsOff0]);
            async_cp16(W + bRow1 + kn, &Bs[nbuf][ldsOff1]);
#else
            ra0 = *(const uint4*)(A + aRow0 + kn);
            ra1 = *(const uint4*)(A + aRow1 + kn);
            rb0 = *(const uint4*)(W + bRow0 + kn);
            rb1 = *(const uint4*)(W + bRow1 + kn);
#endif
        }

        // ---- compute current stage: 2 A-frags x 4 B-frags = 8 WMMA ----
        {
            const unsigned int* As32 = (const unsigned int*)As[buf];
            const unsigned int* Bs32 = (const unsigned int*)Bs[buf];
            const int ar0 = (wave_m * 32 + r15) * 16;
            const int ar1 = ar0 + 16 * 16;
            Frag a0, a1;
#pragma unroll
            for (int v = 0; v < 8; ++v) {
                int p = (v & 3) + ((v >= 4) ? 8 : 0) + phi;
                a0.u[v] = As32[ar0 + p];
                a1.u[v] = As32[ar1 + p];
            }
            Frag b[4];
#pragma unroll
            for (int j = 0; j < 4; ++j) {
                const int br = (wave_n * 64 + j * 16 + r15) * 16;
#pragma unroll
                for (int v = 0; v < 8; ++v)
                    b[j].u[v] = Bs32[br + (v & 3) + ((v >= 4) ? 8 : 0) + phi];
            }
#pragma unroll
            for (int j = 0; j < 4; ++j) {
                acc[0][j] = __builtin_amdgcn_wmma_f32_16x16x32_bf16(
                    false, a0.v, false, b[j].v, (short)0, acc[0][j], false, false);
                acc[1][j] = __builtin_amdgcn_wmma_f32_16x16x32_bf16(
                    false, a1.v, false, b[j].v, (short)0, acc[1][j], false, false);
            }
        }

        // ---- publish next stage ----
        if (more) {
#if HAVE_ASYNC_LDS
            wait_async0();
#else
            *(uint4*)(&As[nbuf][ldsOff0]) = ra0;
            *(uint4*)(&As[nbuf][ldsOff1]) = ra1;
            *(uint4*)(&Bs[nbuf][ldsOff0]) = rb0;
            *(uint4*)(&Bs[nbuf][ldsOff1]) = rb1;
#endif
        }
        __syncthreads();
        buf = nbuf;
    }

    // ---------------- epilogue ----------------
    const int m_in = ((lane & 16) ? 8 : 0);
#pragma unroll
    for (int i2 = 0; i2 < 2; ++i2)
#pragma unroll
        for (int j = 0; j < 4; ++j)
#pragma unroll
            for (int v = 0; v < 8; ++v) {
                int gm = m0 + wave_m * 32 + i2 * 16 + v + m_in;
                int gn = n0 + wave_n * 64 + j * 16 + r15;
                float val = acc[i2][j][v] + bias[gn];
                if (EPI == 0) {
                    int bi = gm / L, s = gm % L;
                    int h = gn >> 7, d = gn & 127;
                    ((unsigned short*)out)[(((size_t)(bi * NHH + h) * L + s) << 7) + d] = f2bf(val);
                } else if (EPI == 2) {
                    int bi = gm / L, t = gm % L;
                    int h = gn >> 7, d = gn & 127;
                    ((unsigned short*)out)[(((size_t)(bi * NHH + h) << 7) + d) * (size_t)L + t] = f2bf(val);
                } else {
                    size_t idx = (size_t)gm * N + gn;
                    ((float*)out)[idx] = val + n2n(resid[idx]);
                }
            }
}

// ---------------- kernel 3: attention (one wave per 16 S-rows) ----------------
__global__ __launch_bounds__(32) void attn_kernel(
    const unsigned short* __restrict__ qb,   // [B,NH,S,HD]
    const unsigned short* __restrict__ kb,   // [B,NH,T,HD]
    const unsigned short* __restrict__ vtb,  // [B,NH,HD,T]
    const float* __restrict__ mask,          // [B,T]
    unsigned short* __restrict__ ctxb) {     // [B,S,H] bf16

    __shared__ float sc[16 * TT];            // 16 KB score tile
    __shared__ unsigned short pr[16 * TT];   // 8 KB bf16 probs

    const int lane = threadIdx.x;
    const int bid  = blockIdx.x;
    const int bh   = bid >> 7;               // S/16 = 128 tiles per (b,h)
    const int s0   = (bid & 127) << 4;
    const int b    = bh >> 4;
    const int h    = bh & 15;

    const int phi = (lane & 16) ? 4 : 0;
    const int r15 = lane & 15;

    // ---- scores: 16 x 256 via 4 K-steps x 16 N-tiles of WMMA ----
    v8f acc[16];
#pragma unroll
    for (int i = 0; i < 16; ++i) acc[i] = (v8f){};

    const unsigned int* qrow = (const unsigned int*)(qb + ((size_t)bh * SS + s0) * HD);
    const unsigned int* kbase = (const unsigned int*)(kb + (size_t)bh * TT * HD);

#pragma unroll
    for (int kk = 0; kk < 4; ++kk) {
        const int kh = kk * 16;              // k0/2 in uints
        Frag a;
#pragma unroll
        for (int v = 0; v < 8; ++v)
            a.u[v] = qrow[r15 * 64 + kh + (v & 3) + ((v >= 4) ? 8 : 0) + phi];
#pragma unroll
        for (int nt = 0; nt < 16; ++nt) {
            Frag bf;
            const int t = nt * 16 + r15;
#pragma unroll
            for (int v = 0; v < 8; ++v)
                bf.u[v] = kbase[t * 64 + kh + (v & 3) + ((v >= 4) ? 8 : 0) + phi];
            acc[nt] = __builtin_amdgcn_wmma_f32_16x16x32_bf16(false, a.v, false, bf.v,
                                                              (short)0, acc[nt], false, false);
        }
    }

    // spill scores to LDS
    const int m_in = ((lane & 16) ? 8 : 0);
#pragma unroll
    for (int nt = 0; nt < 16; ++nt)
#pragma unroll
        for (int v = 0; v < 8; ++v)
            sc[(v + m_in) * TT + nt * 16 + r15] = acc[nt][v];
    __syncthreads();

    // ---- softmax over T=256 (lanes 0..15, one row each) ----
    if (lane < 16) {
        const int m = lane;
        const float* mrow = mask + (size_t)b * TT;
        float mx = -1e30f;
        for (int t = 0; t < TT; ++t) {
            float x = sc[m * TT + t] * SCALE_INV;
            x = fminf(fmaxf(x, -50.0f), 50.0f);
            x += mrow[t] * -1e9f;
            sc[m * TT + t] = x;
            mx = fmaxf(mx, x);
        }
        float sum = 0.0f;
        for (int t = 0; t < TT; ++t) {
            float e = __expf(sc[m * TT + t] - mx);
            sc[m * TT + t] = e;
            sum += e;
        }
        float inv = 1.0f / (sum + 1e-8f);
        for (int t = 0; t < TT; ++t)
            pr[m * TT + t] = f2bf(sc[m * TT + t] * inv);
    }
    __syncthreads();

    // ---- ctx = probs @ V : 16 x 128, K over T=256 ----
    v8f c[8];
#pragma unroll
    for (int i = 0; i < 8; ++i) c[i] = (v8f){};

    const unsigned int* prow  = (const unsigned int*)pr;                 // 128 uints/row
    const unsigned int* vbase = (const unsigned int*)(vtb + (size_t)bh * HD * TT);

#pragma unroll
    for (int kk = 0; kk < 8; ++kk) {
        const int kh = kk * 16;
        Frag a;
#pragma unroll
        for (int v = 0; v < 8; ++v)
            a.u[v] = prow[r15 * 128 + kh + (v & 3) + ((v >= 4) ? 8 : 0) + phi];
#pragma unroll
        for (int nt = 0; nt < 8; ++nt) {
            Frag bf;
            const int d = nt * 16 + r15;
#pragma unroll
            for (int v = 0; v < 8; ++v)
                bf.u[v] = vbase[d * 128 + kh + (v & 3) + ((v >= 4) ? 8 : 0) + phi];
            c[nt] = __builtin_amdgcn_wmma_f32_16x16x32_bf16(false, a.v, false, bf.v,
                                                            (short)0, c[nt], false, false);
        }
    }

    // ctx -> [B,S,H] bf16
#pragma unroll
    for (int nt = 0; nt < 8; ++nt)
#pragma unroll
        for (int v = 0; v < 8; ++v) {
            int s = s0 + v + m_in;
            size_t idx = ((size_t)(b * SS + s)) * HH + h * HD + nt * 16 + r15;
            ctxb[idx] = f2bf(c[nt][v]);
        }
}

// ---------------- kernel 4: LayerNorm (one block per row) ----------------
__global__ __launch_bounds__(256) void layernorm_kernel(const float* __restrict__ x,
                                                        const float* __restrict__ g,
                                                        const float* __restrict__ bta,
                                                        float* __restrict__ out) {
    __shared__ float rs[256], rq[256];
    const int tid = threadIdx.x;
    const size_t row = (size_t)blockIdx.x * HH;

    float s = 0.0f, q = 0.0f;
#pragma unroll
    for (int j = 0; j < HH / 256; ++j) {
        float v = x[row + tid + j * 256];
        s += v;
        q += v * v;
    }
    rs[tid] = s;
    rq[tid] = q;
    __syncthreads();
    for (int off = 128; off > 0; off >>= 1) {
        if (tid < off) { rs[tid] += rs[tid + off]; rq[tid] += rq[tid + off]; }
        __syncthreads();
    }
    const float mu  = rs[0] * (1.0f / HH);
    const float var = rq[0] * (1.0f / HH) - mu * mu;
    const float r   = rsqrtf(var + LN_EPS);
#pragma unroll
    for (int j = 0; j < HH / 256; ++j) {
        int i = tid + j * 256;
        out[row + i] = (x[row + i] - mu) * r * g[i] + bta[i];
    }
}

// ---------------- launcher ----------------
extern "C" void kernel_launch(void* const* d_in, const int* in_sizes, int n_in,
                              void* d_out, int out_size, void* d_ws, size_t ws_size,
                              hipStream_t stream) {
    const float* hs   = (const float*)d_in[0];   // (B,S,H)
    const float* at   = (const float*)d_in[1];   // (B,T,H)
    const float* mask = (const float*)d_in[2];   // (B,T)
    const float* Wq   = (const float*)d_in[3];
    const float* bq   = (const float*)d_in[4];
    const float* Wk   = (const float*)d_in[5];
    const float* bk   = (const float*)d_in[6];
    const float* Wv   = (const float*)d_in[7];
    const float* bv   = (const float*)d_in[8];
    const float* Wo   = (const float*)d_in[9];
    const float* bo   = (const float*)d_in[10];
    const float* ln_g = (const float*)d_in[11];
    const float* ln_b = (const float*)d_in[12];
    float* outp = (float*)d_out;

    const size_t BS = (size_t)BB * SS;   // 8192
    const size_t BT = (size_t)BB * TT;   // 1024

    // workspace carve-up (bf16 stored as ushort)
    char* w = (char*)d_ws;
    unsigned short* hsb  = (unsigned short*)w;                  w += BS * HH * 2;
    unsigned short* atb  = (unsigned short*)w;                  w += BT * HH * 2;
    unsigned short* wqb  = (unsigned short*)w;                  w += (size_t)HH * HH * 2;
    unsigned short* wkb  = (unsigned short*)w;                  w += (size_t)HH * HH * 2;
    unsigned short* wvb  = (unsigned short*)w;                  w += (size_t)HH * HH * 2;
    unsigned short* wob  = (unsigned short*)w;                  w += (size_t)HH * HH * 2;
    unsigned short* qb   = (unsigned short*)w;                  w += BS * HH * 2;     // [B,NH,S,HD]
    unsigned short* kbuf = (unsigned short*)w;                  w += BT * HH * 2;     // [B,NH,T,HD]
    unsigned short* vtb  = (unsigned short*)w;                  w += BT * HH * 2;     // [B,NH,HD,T]
    unsigned short* ctxb = (unsigned short*)w;                  w += BS * HH * 2;     // [B,S,H]
    float*          opre = (float*)w;                           w += BS * HH * 4;

    // 1) converts
    convert_bf16_kernel<<<4096, 256, 0, stream>>>(hs, hsb, (long)(BS * HH));
    convert_bf16_kernel<<<2048, 256, 0, stream>>>(at, atb, (long)(BT * HH));
    convert_bf16_kernel<<<2048, 256, 0, stream>>>(Wq, wqb, (long)HH * HH);
    convert_bf16_kernel<<<2048, 256, 0, stream>>>(Wk, wkb, (long)HH * HH);
    convert_bf16_kernel<<<2048, 256, 0, stream>>>(Wv, wvb, (long)HH * HH);
    convert_bf16_kernel<<<2048, 256, 0, stream>>>(Wo, wob, (long)HH * HH);

    // 2) projections (block tile 128x128)
    gemm_bf16_kernel<0><<<dim3(HH / 128, BS / 128), 256, 0, stream>>>(
        hsb, wqb, bq, qb, nullptr, (int)BS, HH, HH, SS);
    gemm_bf16_kernel<0><<<dim3(HH / 128, BT / 128), 256, 0, stream>>>(
        atb, wkb, bk, kbuf, nullptr, (int)BT, HH, HH, TT);
    gemm_bf16_kernel<2><<<dim3(HH / 128, BT / 128), 256, 0, stream>>>(
        atb, wvb, bv, vtb, nullptr, (int)BT, HH, HH, TT);

    // 3) attention: one wave per 16 S-rows per head
    attn_kernel<<<BB * NHH * (SS / 16), 32, 0, stream>>>(qb, kbuf, vtb, mask, ctxb);

    // 4) output projection + bias + residual (f32)
    gemm_bf16_kernel<3><<<dim3(HH / 128, BS / 128), 256, 0, stream>>>(
        ctxb, wob, bo, opre, hs, (int)BS, HH, HH, SS);

    // 5) LayerNorm -> d_out
    layernorm_kernel<<<(int)BS, 256, 0, stream>>>(opre, ln_g, ln_b, outp);
}
